// ConvNeXTBlock_88854283420005
// MI455X (gfx1250) — matrix operand order repaired and checked by reading, not compiled, not run on hardware
//
#include <hip/hip_runtime.h>
#include <math.h>

#define C_    128
#define HID_  512
#define KW    7
#define PADW  3
#define B_    4
#define T_    512
#define F_    128
#define TF_   (T_ * F_)
#define NPTS  (B_ * T_ * F_)

typedef _Float16 v16h __attribute__((ext_vector_type(16)));
typedef float    v8f  __attribute__((ext_vector_type(8)));

union F16Frag { v16h v; uint4 q[2]; };
union Pack8   { _Float16 h[8]; uint4 q; };

__device__ __forceinline__ float tanh_fast(float u) {
#if __has_builtin(__builtin_amdgcn_tanhf)
  return __builtin_amdgcn_tanhf(u);          // v_tanh_f32 (CDNA5 TRANS op)
#elif __has_builtin(__builtin_amdgcn_tanh_f32)
  return __builtin_amdgcn_tanh_f32(u);
#else
  return tanhf(u);
#endif
}

__device__ __forceinline__ float gelu_tanh(float v) {
  float u = 0.7978845608028654f * (v + 0.044715f * v * v * v);
  return 0.5f * v * (1.0f + tanh_fast(u));
}

// ---------------------------------------------------------------------------
// Kernel 1: depthwise 7x7 conv + bias, fused LayerNorm over C, emit f16
// channels-last activations [B,T,F,C]. Double-buffered halo rows: one
// barrier per channel, next channel's global loads overlap current compute.
// ---------------------------------------------------------------------------
__global__ __launch_bounds__(128)
void conv_ln_kernel(const float* __restrict__ x,
                    const float* __restrict__ dw_w,
                    const float* __restrict__ dw_b,
                    const float* __restrict__ ln_g,
                    const float* __restrict__ ln_b,
                    _Float16* __restrict__ yln)
{
  __shared__ float    srow[2][KW][F_ + 8];  // double-buffered halo rows
  __shared__ _Float16 lnbuf[F_][C_ + 8];    // per-point channel vector (f16)

  const int f = threadIdx.x;
  const int t = blockIdx.x;
  const int b = blockIdx.y;

  // zero the F-halo columns once (always zero: out-of-image padding)
  if (f < PADW) {
    #pragma unroll
    for (int d = 0; d < 2; ++d)
      #pragma unroll
      for (int rr = 0; rr < KW; ++rr) {
        srow[d][rr][f] = 0.f;
        srow[d][rr][F_ + PADW + f] = 0.f;
      }
  }

  const float* xbase = x + (size_t)b * C_ * TF_;

  // preload channel 0 rows into buffer 0
  #pragma unroll
  for (int rr = 0; rr < KW; ++rr) {
    int tt = t + rr - PADW;
    float v = 0.f;
    if (tt >= 0 && tt < T_) v = xbase[(size_t)0 * TF_ + tt * F_ + f];
    srow[0][rr][f + PADW] = v;
  }

  float sum = 0.f, sumsq = 0.f;             // fp32 LN statistics

  for (int c = 0; c < C_; ++c) {
    const int cur = c & 1;
    __syncthreads();                        // buf[cur] loads + prior reads done
    if (c + 1 < C_) {                       // prefetch next channel's rows
      const float* xp = xbase + (size_t)(c + 1) * TF_;
      #pragma unroll
      for (int rr = 0; rr < KW; ++rr) {
        int tt = t + rr - PADW;
        float v = 0.f;
        if (tt >= 0 && tt < T_) v = xp[tt * F_ + f];
        srow[cur ^ 1][rr][f + PADW] = v;
      }
    }

    const float* wc = dw_w + c * (KW * KW);
    float acc = dw_b[c];
    #pragma unroll
    for (int kt = 0; kt < KW; ++kt) {
      #pragma unroll
      for (int kf = 0; kf < KW; ++kf) {
        acc += srow[cur][kt][f + kf] * wc[kt * KW + kf];
      }
    }
    lnbuf[f][c] = (_Float16)acc;            // thread f owns row f: no barrier
    sum   += acc;
    sumsq += acc * acc;
  }

  const float mean = sum * (1.0f / C_);
  const float var  = sumsq * (1.0f / C_) - mean * mean;
  const float rstd = rsqrtf(var + 1e-5f);

  _Float16* dst = yln + ((size_t)((b * T_ + t) * F_ + f)) * C_;
  for (int c0 = 0; c0 < C_; c0 += 8) {
    Pack8 p;
    #pragma unroll
    for (int j = 0; j < 8; ++j) {
      int c = c0 + j;
      float v = ((float)lnbuf[f][c] - mean) * rstd * ln_g[c] + ln_b[c];
      p.h[j] = (_Float16)v;
    }
    *(uint4*)(dst + c0) = p.q;              // 16B packed f16 store
  }
}

// ---------------------------------------------------------------------------
// Kernel 2: pointwise MLP (128 -> 512 -> GELU -> 512 -> 128) + LayerScale.
// WMMA K-chains start from an inline-0 C operand; biases / LayerScale are
// folded into epilogues (off the WMMA critical path). Bias/scale scalars are
// loaded per use (not hoisted) to keep allocation under 256 VGPRs and avoid
// s_set_vgpr_msb traffic.
// ---------------------------------------------------------------------------
#define SMEM_W1   (HID_ * C_ * 2)                 // 131072 B
#define SMEM_W2   (C_ * HID_ * 2)                 // 131072 B
#define SMEM_BIAS ((HID_ + C_ + C_) * 4)          // 3072 B
#define SMEM_SCR  (8 * 4096)                      // per-wave scratch
#define SMEM_TOTAL (SMEM_W1 + SMEM_W2 + SMEM_BIAS + SMEM_SCR)  // 297984 B

__global__ __launch_bounds__(256)
void mlp_wmma_kernel(const _Float16* __restrict__ yln,
                     const float* __restrict__ w1, const float* __restrict__ b1,
                     const float* __restrict__ w2, const float* __restrict__ b2,
                     const float* __restrict__ ls,
                     float* __restrict__ out)
{
  extern __shared__ __align__(16) char smem[];
  _Float16* sW1 = (_Float16*)smem;                    // [HID][C] row-major
  _Float16* sW2 = (_Float16*)(smem + SMEM_W1);        // [C][HID] row-major
  float*    sB1 = (float*)(smem + SMEM_W1 + SMEM_W2);
  float*    sB2 = sB1 + HID_;
  float*    sLS = sB2 + C_;
  char*     scrBase = (char*)(sLS + C_);

  const int tid = threadIdx.x;
  for (int i = tid; i < HID_ * C_; i += 256) sW1[i] = (_Float16)w1[i];
  for (int i = tid; i < C_ * HID_; i += 256) sW2[i] = (_Float16)w2[i];
  for (int i = tid; i < HID_; i += 256) sB1[i] = b1[i];
  if (tid < C_) { sB2[tid] = b2[tid]; sLS[tid] = ls[tid]; }
  __syncthreads();

  const int lane = tid & 31;
  const int wave = tid >> 5;
  const int r    = lane & 15;   // col index for B/C/D frags, row for A frags
  const int hi   = lane >> 4;   // lane-half select per ISA fragment layout

  _Float16* scrH = (_Float16*)(scrBase + wave * 4096); // 16x32 f16 transpose
  float*    scrF = (float*)scrH;                       // 64x16 f32 out staging

  const int ntiles = NPTS / 16;
  for (int tile = blockIdx.x * 8 + wave; tile < ntiles; tile += gridDim.x * 8) {
    const int p0 = tile * 16;

    // --- A1 fragments: 16 points x 128 channels, fp16, in registers --------
    F16Frag a1[4];
    const _Float16* arow = yln + (size_t)(p0 + r) * C_;
    #pragma unroll
    for (int ks = 0; ks < 4; ++ks) {
      a1[ks].q[0] = *(const uint4*)(arow + ks * 32 + hi * 8);
      a1[ks].q[1] = *(const uint4*)(arow + ks * 32 + 16 + hi * 8);
    }

    // --- phase 1: hidden activations as 8 A2 fragments in registers --------
    F16Frag a2[8];
    #pragma unroll
    for (int pair = 0; pair < 8; ++pair) {
      #pragma unroll
      for (int j = 0; j < 2; ++j) {
        const int hb = (pair * 2 + j) * 16;
        const float bb1 = sB1[hb + r];      // consumed only in GELU epilogue
        const _Float16* wrow = sW1 + (size_t)(hb + r) * C_ + hi * 8;
        F16Frag bf[2];
        bf[0].q[0] = *(const uint4*)(wrow);
        bf[0].q[1] = *(const uint4*)(wrow + 16);
        v8f hacc = {};                      // inline-0 C operand on first WMMA
        #pragma unroll
        for (int ks = 0; ks < 4; ++ks) {
          if (ks < 3) {                     // double-buffer next B1 fragment
            bf[(ks + 1) & 1].q[0] = *(const uint4*)(wrow + (ks + 1) * 32);
            bf[(ks + 1) & 1].q[1] = *(const uint4*)(wrow + (ks + 1) * 32 + 16);
          }
          hacc = __builtin_amdgcn_wmma_f32_16x16x32_f16(
              false, a1[ks].v, false, bf[ks & 1].v, (short)0, hacc,
              false, false);
        }
        // bias + GELU, downcast, transpose 16x16 chunk into per-wave scratch
        #pragma unroll
        for (int i = 0; i < 8; ++i) {
          float g = gelu_tanh(hacc[i] + bb1);
          scrH[(i + hi * 8) * 32 + j * 16 + r] = (_Float16)g;
        }
      }
      // A2 fragment (16 points x 32 hidden) from transposed scratch
      a2[pair].q[0] = *(const uint4*)(scrH + r * 32 + hi * 8);
      a2[pair].q[1] = *(const uint4*)(scrH + r * 32 + 16 + hi * 8);
    }

    // --- phase 2: one output channel-tile at a time -------------------------
    const int b   = p0 / TF_;
    const int rem = p0 - b * TF_;                 // t*F + f0 (tile-aligned)
    float* obase = out + (size_t)b * C_ * TF_ + rem;
    #pragma unroll
    for (int q = 0; q < 2; ++q) {
      for (int n2s = 0; n2s < 4; ++n2s) {
        const int n2 = q * 4 + n2s;
        const float s  = sLS[n2 * 16 + r];  // consumed only in staging FMA
        const float bl = sB2[n2 * 16 + r] * s;
        const _Float16* w2row = sW2 + (size_t)(n2 * 16 + r) * HID_ + hi * 8;
        F16Frag bf[2];
        bf[0].q[0] = *(const uint4*)(w2row);
        bf[0].q[1] = *(const uint4*)(w2row + 16);
        v8f oacc = {};                      // inline-0 C operand on first WMMA
        #pragma unroll
        for (int pair = 0; pair < 8; ++pair) {
          if (pair < 7) {                   // double-buffer next B2 fragment
            bf[(pair + 1) & 1].q[0] = *(const uint4*)(w2row + (pair + 1) * 32);
            bf[(pair + 1) & 1].q[1] =
                *(const uint4*)(w2row + (pair + 1) * 32 + 16);
          }
          oacc = __builtin_amdgcn_wmma_f32_16x16x32_f16(
              false, a2[pair].v, false, bf[pair & 1].v, (short)0, oacc,
              false, false);
        }
        // (sum + b2)*ls == sum*ls + b2*ls : one v_fma per element in staging
        #pragma unroll
        for (int i = 0; i < 8; ++i)
          scrF[(n2s * 16 + r) * 16 + i + hi * 8] = fmaf(oacc[i], s, bl);
      }
      #pragma unroll
      for (int cc = 0; cc < 2; ++cc) {
        int cl = lane + cc * 32;                  // 0..63 local channel
        int c  = q * 64 + cl;
        const float4* src = (const float4*)(scrF + cl * 16);
        float4* dst = (float4*)(obase + (size_t)c * TF_);
        dst[0] = src[0]; dst[1] = src[1]; dst[2] = src[2]; dst[3] = src[3];
      }
    }
  }
}

// ---------------------------------------------------------------------------
extern "C" void kernel_launch(void* const* d_in, const int* in_sizes, int n_in,
                              void* d_out, int out_size, void* d_ws, size_t ws_size,
                              hipStream_t stream) {
  const float* x    = (const float*)d_in[0];
  const float* dw_w = (const float*)d_in[1];
  const float* dw_b = (const float*)d_in[2];
  const float* ln_g = (const float*)d_in[3];
  const float* ln_b = (const float*)d_in[4];
  const float* w1   = (const float*)d_in[5];
  const float* b1   = (const float*)d_in[6];
  const float* w2   = (const float*)d_in[7];
  const float* b2   = (const float*)d_in[8];
  const float* ls   = (const float*)d_in[9];

  _Float16* yln = (_Float16*)d_ws;   // [B,T,F,C] f16, 64 MiB

  conv_ln_kernel<<<dim3(T_, B_), 128, 0, stream>>>(x, dw_w, dw_b, ln_g, ln_b, yln);
  mlp_wmma_kernel<<<dim3(512), 256, SMEM_TOTAL, stream>>>(
      yln, w1, b1, w2, b2, ls, (float*)d_out);
}
